// LSTMhead_33681133536064
// MI455X (gfx1250) — matrix operand order repaired and checked by reading, not compiled, not run on hardware
//
#include <hip/hip_runtime.h>

// ---------------------------------------------------------------------------
// CDNA5 (gfx1250) wave32 WMMA implementation of the CenterNet LSTM head.
// All matrix math routed through v_wmma_f32_16x16x32_bf16.
// ---------------------------------------------------------------------------

typedef __attribute__((ext_vector_type(16))) __bf16 v16bf;
typedef __attribute__((ext_vector_type(8)))  float  v8f;

// round-to-nearest-even f32 -> bf16
__device__ __forceinline__ __bf16 f2bf(float f) {
  unsigned u = __builtin_bit_cast(unsigned, f);
  unsigned r = (u + 0x7FFFu + ((u >> 16) & 1u)) >> 16;
  unsigned short s = (unsigned short)r;
  return __builtin_bit_cast(__bf16, s);
}

__device__ __forceinline__ v8f vzero8() {
  v8f z;
#pragma unroll
  for (int i = 0; i < 8; ++i) z[i] = 0.f;
  return z;
}

__device__ __forceinline__ v8f wmma_bf16(v16bf a, v16bf b, v8f c) {
  // (neg_a, A, neg_b, B, c_mod, C, reuse_a, reuse_b)
  return __builtin_amdgcn_wmma_f32_16x16x32_bf16(false, a, false, b, (short)0, c,
                                                 false, false);
}

// A fragment (16x32, bf16): lane holds row M=lane%16.
// K(lane,e) = kbase + 8*(lane/16) + (e<8 ? e : e+8)   [per ISA 16-bit A layout]
__device__ __forceinline__ v16bf afrag_f32(const float* src, int ld, int kbase,
                                           int lane) {
  v16bf a;
  const float* row = src + (size_t)(lane & 15) * ld + kbase + ((lane >> 4) << 3);
#pragma unroll
  for (int e = 0; e < 8; ++e) {
    a[e] = f2bf(row[e]);
    a[8 + e] = f2bf(row[16 + e]);
  }
  return a;
}

__device__ __forceinline__ v16bf afrag_lds(const __bf16* src, int ld, int kbase,
                                           int lane) {
  v16bf a;
  const __bf16* row = src + (lane & 15) * ld + kbase + ((lane >> 4) << 3);
#pragma unroll
  for (int e = 0; e < 8; ++e) {
    a[e] = row[e];
    a[8 + e] = row[16 + e];
  }
  return a;
}

// B fragment (32x16 KxN) from weights stored [N][K] row-major:
// lane: N = n0 + lane%16, K = kbase + 8*(lane/16) + (e<8 ? e : e+8)
__device__ __forceinline__ v16bf bfrag_f32(const float* Wnk, int n0, int ld,
                                           int kbase, int lane) {
  v16bf b;
  const float* row =
      Wnk + (size_t)(n0 + (lane & 15)) * ld + kbase + ((lane >> 4) << 3);
#pragma unroll
  for (int e = 0; e < 8; ++e) {
    b[e] = f2bf(row[e]);
    b[8 + e] = f2bf(row[16 + e]);
  }
  return b;
}

__device__ __forceinline__ v16bf bfrag_lds(const __bf16* Wnk, int n0, int ld,
                                           int kbase, int lane) {
  v16bf b;
  const __bf16* row = Wnk + (n0 + (lane & 15)) * ld + kbase + ((lane >> 4) << 3);
#pragma unroll
  for (int e = 0; e < 8; ++e) {
    b[e] = row[e];
    b[8 + e] = row[16 + e];
  }
  return b;
}

// ---------------------------------------------------------------------------
// Kernel 1: fused conv3x3(64->64)+bias+relu then conv1x1(64->32)+bias.
// One block per (image b, row y). 8 waves, each owns a 16-pixel segment.
// Implicit GEMM1: [16 x 576] x [576 x 64], K mapped as k = tap*64 + ch so a
// lane's A fragment is two contiguous 16B LDS runs (ds_load_b128 x2).
// GEMM2: [16 x 64] x [64 x 32].
// ---------------------------------------------------------------------------
__global__ void __launch_bounds__(256) conv_fused_kernel(
    const float* __restrict__ x, const float* __restrict__ w1,
    const float* __restrict__ b1, const float* __restrict__ w2,
    const float* __restrict__ b2, float* __restrict__ feat) {
  extern __shared__ char smem[];
  __bf16* in_lds = (__bf16*)smem;          // [3 dy][132 xp][64 ch] (ch fastest)
  __bf16* w1_lds = in_lds + 3 * 132 * 64;  // [64 n][576 k], k = tap*64 + ch
  __bf16* w2_lds = w1_lds + 64 * 576;      // [32 n][64 k]
  __bf16* mid_lds = w2_lds + 32 * 64;      // [8 waves][16 m][64 n]

  const int tid = threadIdx.x;
  const int b = blockIdx.x >> 7;
  const int y = blockIdx.x & 127;

  // stage conv1 weights permuted to k = tap*64 + ch
  for (int i = tid; i < 64 * 576; i += 256) {
    int n = i / 576, k = i - n * 576;
    int tap = k >> 6, ch = k & 63;
    w1_lds[i] = f2bf(w1[n * 576 + ch * 9 + tap]);
  }
  for (int i = tid; i < 32 * 64; i += 256) w2_lds[i] = f2bf(w2[i]);
  // stage input rows y-1..y+1 (coalesced global read over xp, scatter to LDS
  // channel-fastest layout), zero-padded columns xp=0 and xp>=129.
  for (int i = tid; i < 64 * 3 * 132; i += 256) {
    int ch = i / 396, r = i - ch * 396, dy = r / 132, xp = r - dy * 132;
    int yy = y + dy - 1, xi = xp - 1;
    float v = 0.f;
    if (yy >= 0 && yy < 128 && xi >= 0 && xi < 128)
      v = x[(((size_t)b * 64 + ch) << 14) + (yy << 7) + xi];
    in_lds[(dy * 132 + xp) * 64 + ch] = f2bf(v);
  }
  __syncthreads();

  const int lane = tid & 31, w = tid >> 5;
  const int xb = w << 4;
  const int xm = xb + (lane & 15);  // this lane's pixel column (A row)

  // GEMM1: K = 576 = 18 k-steps of 32
  v8f acc[4];
#pragma unroll
  for (int t = 0; t < 4; ++t) acc[t] = vzero8();
  for (int ks = 0; ks < 18; ++ks) {
    // k0 % 64 in {0,8,32,40}: both 8-element runs (k0.. and k0+16..) stay in
    // one tap block -> single (dy,dx) and two contiguous 16B LDS loads.
    int k0 = (ks << 5) + ((lane >> 4) << 3);
    int tap = k0 >> 6, cb = k0 & 63;
    int dy = tap / 3, dx = tap - dy * 3;
    const __bf16* p = in_lds + (dy * 132 + xm + dx) * 64 + cb;
    v16bf a;
#pragma unroll
    for (int e = 0; e < 8; ++e) {
      a[e] = p[e];
      a[8 + e] = p[16 + e];
    }
#pragma unroll
    for (int nt = 0; nt < 4; ++nt) {
      v16bf bb = bfrag_lds(w1_lds, nt << 4, 576, ks << 5, lane);
      acc[nt] = wmma_bf16(a, bb, acc[nt]);
    }
  }
  // bias + relu -> mid (bf16) for second GEMM. C layout: M = v + 8*(lane/16).
  __bf16* mymid = mid_lds + (w << 10);
#pragma unroll
  for (int nt = 0; nt < 4; ++nt) {
    int n = (nt << 4) + (lane & 15);
    int mbase = (lane >> 4) << 3;
    float bias = b1[n];
#pragma unroll
    for (int v = 0; v < 8; ++v) {
      float val = acc[nt][v] + bias;
      mymid[((mbase + v) << 6) + n] = f2bf(val > 0.f ? val : 0.f);
    }
  }
  __syncthreads();

  // GEMM2: [16 x 64] x [64 x 32]
  v8f acc2[2];
  acc2[0] = vzero8();
  acc2[1] = vzero8();
#pragma unroll
  for (int ks = 0; ks < 2; ++ks) {
    v16bf a = afrag_lds(mymid, 64, ks << 5, lane);
#pragma unroll
    for (int nt = 0; nt < 2; ++nt) {
      v16bf bb = bfrag_lds(w2_lds, nt << 4, 64, ks << 5, lane);
      acc2[nt] = wmma_bf16(a, bb, acc2[nt]);
    }
  }
#pragma unroll
  for (int nt = 0; nt < 2; ++nt) {
    int n = (nt << 4) + (lane & 15);
    float bias = b2[n];
    int mbase = (lane >> 4) << 3;
#pragma unroll
    for (int v = 0; v < 8; ++v)
      feat[(((size_t)b * 32 + n) << 14) + (y << 7) + xb + mbase + v] =
          acc2[nt][v] + bias;
  }
}

// ---------------------------------------------------------------------------
// Kernel 2: keypoint gather. g[t][bp][d], bp = branch*256 + b*16 + n.
// branch 0: (r,c) = (i0,i1); branch 1: (r,c) = (i1,i0). idx clamped to [0,127].
// ---------------------------------------------------------------------------
__global__ void __launch_bounds__(256) gather_kernel(
    const float* __restrict__ feat, const int* __restrict__ kp,
    float* __restrict__ g) {
  int i = blockIdx.x * 256 + threadIdx.x;  // over 32*512*32
  if (i >= 32 * 512 * 32) return;
  int d = i & 31;
  int bp = (i >> 5) & 511;
  int t = i >> 14;
  int branch = bp >> 8;
  int wbn = bp & 255;  // b*16+n
  const int* ip = kp + ((wbn << 5) + t) * 2;
  int i0 = ip[0], i1 = ip[1];
  i0 = i0 < 0 ? 0 : (i0 > 127 ? 127 : i0);
  i1 = i1 < 0 ? 0 : (i1 > 127 ? 127 : i1);
  int r = branch == 0 ? i0 : i1;
  int c = branch == 0 ? i1 : i0;
  g[i] = feat[(((size_t)(wbn >> 4) * 32 + d) << 14) + (r << 7) + c];
}

// ---------------------------------------------------------------------------
// Kernel 3: generic WMMA GEMM  C[M][N] = A[M][K] @ W[N][K]^T + bias1 + bias2,
// optional ReLU. Block = 16 rows (one m-tile), 8 waves split the n-tiles.
// ---------------------------------------------------------------------------
__global__ void __launch_bounds__(256) gemm_bias_kernel(
    const float* __restrict__ A, const float* __restrict__ W,
    const float* __restrict__ bias1, const float* __restrict__ bias2,
    float* __restrict__ C, int K, int N, int relu) {
  const int lane = threadIdx.x & 31, w = threadIdx.x >> 5;
  const int m0 = blockIdx.x << 4;
  const int ksteps = K >> 5;  // K in {32, 128}

  v16bf af[4];
  for (int ks = 0; ks < ksteps; ++ks)
    af[ks] = afrag_f32(A + (size_t)m0 * K, K, ks << 5, lane);

  const int ntiles = N >> 4;
  for (int nt = w; nt < ntiles; nt += 8) {
    __builtin_prefetch(W + (size_t)(nt << 4) * K, 0, 1);
    v8f acc = vzero8();
    for (int ks = 0; ks < ksteps; ++ks) {
      v16bf bb = bfrag_f32(W, nt << 4, K, ks << 5, lane);
      acc = wmma_bf16(af[ks], bb, acc);
    }
    int n = (nt << 4) + (lane & 15);
    float bsum = (bias1 ? bias1[n] : 0.f) + (bias2 ? bias2[n] : 0.f);
    int mbase = (lane >> 4) << 3;
#pragma unroll
    for (int v = 0; v < 8; ++v) {
      float val = acc[v] + bsum;
      if (relu) val = val > 0.f ? val : 0.f;
      C[(size_t)(m0 + mbase + v) * N + n] = val;
    }
  }
}

// ---------------------------------------------------------------------------
// Kernel 4: LSTM recurrence. One block owns 16 sequences (32 blocks x 512 seq).
// Whh (512x128 bf16 = 128KB) resident in LDS for all 32 timesteps.
// Per step: gates[16x512] = Xih[t] + h @ Whh^T via 128 WMMAs, then pointwise.
// ---------------------------------------------------------------------------
__global__ void __launch_bounds__(256) lstm_recurrent_kernel(
    const float* __restrict__ xih, const float* __restrict__ whh,
    const float* __restrict__ hc0, float* __restrict__ hc_out,
    float* __restrict__ seq_out) {
  extern __shared__ char smem[];
  __bf16* whh_lds = (__bf16*)smem;            // [512 n][128 k]
  __bf16* h_lds = whh_lds + 512 * 128;        // [16 m][128 u] bf16
  float* c_lds = (float*)(h_lds + 16 * 128);  // [16 m][128 u] f32
  float* g_lds = c_lds + 16 * 128;            // [16 m][512 n] f32

  const int tid = threadIdx.x, lane = tid & 31, w = tid >> 5;
  const int m0 = blockIdx.x << 4;

  for (int i = tid; i < 512 * 128; i += 256) whh_lds[i] = f2bf(whh[i]);
  for (int i = tid; i < 16 * 128; i += 256) {
    int m = i >> 7, u = i & 127;
    float h = hc0 ? hc0[(size_t)(m0 + m) * 128 + u] : 0.f;
    float c = hc0 ? hc0[65536 + (size_t)(m0 + m) * 128 + u] : 0.f;
    h_lds[i] = f2bf(h);
    c_lds[i] = c;
  }
  __syncthreads();

  for (int t = 0; t < 32; ++t) {
    v16bf a4[4];
#pragma unroll
    for (int ks = 0; ks < 4; ++ks) a4[ks] = afrag_lds(h_lds, 128, ks << 5, lane);
#pragma unroll
    for (int q = 0; q < 4; ++q) {
      int nt = (w << 2) + q;
      int n = (nt << 4) + (lane & 15);
      int mbase = (lane >> 4) << 3;
      // seed accumulator with the precomputed input projection (C layout)
      v8f acc;
      const float* xr = xih + ((size_t)(t * 512 + m0) << 9) + n;
#pragma unroll
      for (int v = 0; v < 8; ++v) acc[v] = xr[(size_t)(mbase + v) << 9];
#pragma unroll
      for (int ks = 0; ks < 4; ++ks) {
        v16bf bb = bfrag_lds(whh_lds, nt << 4, 128, ks << 5, lane);
        acc = wmma_bf16(a4[ks], bb, acc);
      }
#pragma unroll
      for (int v = 0; v < 8; ++v) g_lds[((mbase + v) << 9) + n] = acc[v];
    }
    __syncthreads();
    // pointwise LSTM cell (torch gate order i, f, g, o)
    for (int j = tid; j < 16 * 128; j += 256) {
      int m = j >> 7, u = j & 127;
      const float* gm = g_lds + (m << 9);
      float gi = gm[u], gf = gm[128 + u], gg = gm[256 + u], go = gm[384 + u];
      float si = 1.f / (1.f + __expf(-gi));
      float sf = 1.f / (1.f + __expf(-gf));
      float so = 1.f / (1.f + __expf(-go));
      float c = sf * c_lds[j] + si * tanhf(gg);
      float h = so * tanhf(c);
      c_lds[j] = c;
      h_lds[j] = f2bf(h);
      seq_out[((size_t)(t * 512 + m0 + m) << 7) + u] = h;
      if (t == 31 && hc_out) {
        hc_out[(size_t)(m0 + m) * 128 + u] = h;
        hc_out[65536 + (size_t)(m0 + m) * 128 + u] = c;
      }
    }
    __syncthreads();
  }
}

// ---------------------------------------------------------------------------
// Kernel 5: final projection: out = p1(row) . w2 + b2, scattered to [B,N,K,2].
// ---------------------------------------------------------------------------
__global__ void __launch_bounds__(256) proj2_kernel(
    const float* __restrict__ p1, const float* __restrict__ w2,
    const float* __restrict__ b2, float* __restrict__ out) {
  int r = blockIdx.x * 256 + threadIdx.x;  // r = t*512 + bp
  if (r >= 16384) return;
  const float* row = p1 + (size_t)r * 64;
  float s = 0.f;
#pragma unroll
  for (int u = 0; u < 64; ++u) s += row[u] * w2[u];
  s += b2[0];
  int t = r >> 9, bp = r & 511, branch = bp >> 8, wbn = bp & 255;
  out[((wbn << 5) + t) * 2 + branch] = s;
}

// ---------------------------------------------------------------------------
// Kernel 6: gt_keypoints = clamp(idx, 0, 127) * DOWN_SCALE(4)
// ---------------------------------------------------------------------------
__global__ void __launch_bounds__(256) gt_kernel(const int* __restrict__ kp,
                                                 float* __restrict__ out) {
  int i = blockIdx.x * 256 + threadIdx.x;
  if (i >= 16384) return;
  int v = kp[i];
  v = v < 0 ? 0 : (v > 127 ? 127 : v);
  out[16384 + i] = (float)v * 4.0f;
}

// ---------------------------------------------------------------------------
extern "C" void kernel_launch(void* const* d_in, const int* in_sizes, int n_in,
                              void* d_out, int out_size, void* d_ws,
                              size_t ws_size, hipStream_t stream) {
  const float* x = (const float*)d_in[0];
  const int* kp = (const int*)d_in[1];
  const float* conv1_w = (const float*)d_in[2];
  const float* conv1_b = (const float*)d_in[3];
  const float* conv2_w = (const float*)d_in[4];
  const float* conv2_b = (const float*)d_in[5];
  const float* enc_wih = (const float*)d_in[6];
  const float* enc_whh = (const float*)d_in[7];
  const float* enc_bih = (const float*)d_in[8];
  const float* enc_bhh = (const float*)d_in[9];
  const float* dec_wih = (const float*)d_in[10];
  const float* dec_whh = (const float*)d_in[11];
  const float* dec_bih = (const float*)d_in[12];
  const float* dec_bhh = (const float*)d_in[13];
  // px_* (indices 14..17) are dead in the reference forward.
  const float* py_w1 = (const float*)d_in[18];
  const float* py_b1 = (const float*)d_in[19];
  const float* py_w2 = (const float*)d_in[20];
  const float* py_b2 = (const float*)d_in[21];
  float* out = (float*)d_out;

  float* ws = (float*)d_ws;
  float* feat = ws;                    // [16][32][16384]      8388608
  float* g = feat + 8388608;           // [32][512][32]         524288
  float* xih = g + 524288;             // [32*512][512]        8388608 (reused)
  float* enc_out = xih + 8388608;      // [32][512][128]       2097152
  float* dec_out = enc_out + 2097152;  // [32][512][128]       2097152
  float* hc = dec_out + 2097152;       // h[512][128]+c[..]     131072
  float* p1 = hc + 131072;             // [16384][64]          1048576

  size_t conv_lds =
      (size_t)(3 * 132 * 64 + 64 * 576 + 32 * 64 + 8 * 16 * 64) * sizeof(__bf16);
  size_t lstm_lds = (size_t)(512 * 128 + 16 * 128) * sizeof(__bf16) +
                    (size_t)(16 * 128 + 16 * 512) * sizeof(float);

  // 1) fused conv head -> feat
  conv_fused_kernel<<<2048, 256, conv_lds, stream>>>(x, conv1_w, conv1_b,
                                                     conv2_w, conv2_b, feat);
  // 2) keypoint gather (both branches batched: 512 sequences)
  gather_kernel<<<2048, 256, 0, stream>>>(feat, kp, g);
  // 3) encoder input projection: [16384 x 32] @ [32 x 512]^T
  gemm_bias_kernel<<<1024, 256, 0, stream>>>(g, enc_wih, enc_bih, enc_bhh, xih,
                                             32, 512, 0);
  // 4) encoder recurrence
  lstm_recurrent_kernel<<<32, 256, lstm_lds, stream>>>(xih, enc_whh, nullptr,
                                                       hc, enc_out);
  // 5) decoder input projection: [16384 x 128] @ [128 x 512]^T
  gemm_bias_kernel<<<1024, 256, 0, stream>>>(enc_out, dec_wih, dec_bih, dec_bhh,
                                             xih, 128, 512, 0);
  // 6) decoder recurrence (init from encoder final h,c)
  lstm_recurrent_kernel<<<32, 256, lstm_lds, stream>>>(xih, dec_whh, hc,
                                                       nullptr, dec_out);
  // 7) projection layer 1: [16384 x 128] @ [128 x 64]^T + ReLU
  gemm_bias_kernel<<<1024, 256, 0, stream>>>(dec_out, py_w1, py_b1, nullptr, p1,
                                             128, 64, 1);
  // 8) projection layer 2 + scatter into keypoints
  proj2_kernel<<<64, 256, 0, stream>>>(p1, py_w2, py_b2, out);
  // 9) gt_keypoints
  gt_kernel<<<64, 256, 0, stream>>>(kp, out);
}
